// GraphStructured_35837207118465
// MI455X (gfx1250) — matrix operand order repaired and checked by reading, not compiled, not run on hardware
//
#include <hip/hip_runtime.h>
#include <hip/hip_bf16.h>

// CDNA5 / gfx1250. wave32. WMMA bf16 16x16x32 + Tensor Data Mover for LDS staging.

typedef __bf16 bf16_t;
typedef __attribute__((ext_vector_type(16))) __bf16 v16bf;
typedef __attribute__((ext_vector_type(8)))  float   v8f;
typedef __attribute__((ext_vector_type(4)))  unsigned int u32x4;
typedef __attribute__((ext_vector_type(8)))  int         i32x8;
typedef __attribute__((ext_vector_type(4)))  int         i32x4;

#define Bsz   1024
#define NN    32
#define NE    64
#define NDE   128      // directed edges
#define DIMc  768
#define DEPTHc 4
#define OUTD  128

#define HAVE_TDM __has_builtin(__builtin_amdgcn_tensor_load_to_lds)

// ---------------------------------------------------------------- utilities

__global__ void cvt_f32_bf16(const float* __restrict__ src, bf16_t* __restrict__ dst, int n) {
    int i = blockIdx.x * blockDim.x + threadIdx.x;
    if (i < n) dst[i] = (bf16_t)src[i];
}

// one block, 32 threads: incoming-edge CSR (in-degree is 4, bound 8)
__global__ void build_csr(const int* __restrict__ dst, int* __restrict__ incnt,
                          int* __restrict__ inedges) {
    int n = threadIdx.x;
    if (n >= NN) return;
    int c = 0;
    for (int e = 0; e < NDE; ++e)
        if (dst[e] == n && c < 8) inedges[n * 8 + c++] = e;
    incnt[n] = c;
}

// nf[b,n,d] = nfe[n,d] + x[b, n*8 + d%8]   (f32)
// ef[b,e,d] = efe[e,d] + x[b,(32+e)*8+d%8] (bf16)
__global__ void init_feats(const float* __restrict__ x, const float* __restrict__ nfe,
                           const float* __restrict__ efe, float* __restrict__ nf,
                           bf16_t* __restrict__ ef, int total) {
    int idx = blockIdx.x * blockDim.x + threadIdx.x;
    if (idx >= total) return;
    int d = idx % DIMc;
    int e = (idx / DIMc) % (NN + NE);
    int b = idx / (DIMc * (NN + NE));
    float v = x[(size_t)b * DIMc + e * 8 + (d & 7)];
    if (e < NN)
        nf[((size_t)b * NN + e) * DIMc + d] = v + nfe[e * DIMc + d];
    else
        ef[((size_t)b * NE + (e - NN)) * DIMc + d] =
            (bf16_t)(v + efe[(e - NN) * DIMc + d]);
}

// H[b,n,d] = nf[b,n,d] + sum_{e: dst[e]==n} relu(nf[b,src[e],d] + ef[b, e%64, d])
__global__ void msg_agg(const float* __restrict__ nf, const bf16_t* __restrict__ ef,
                        const int* __restrict__ src, const int* __restrict__ incnt,
                        const int* __restrict__ inedges, bf16_t* __restrict__ H, int total) {
    int idx = blockIdx.x * blockDim.x + threadIdx.x;
    if (idx >= total) return;
    int d = idx % DIMc;
    int n = (idx / DIMc) % NN;
    int b = idx / (DIMc * NN);
    const float*  nfb = nf + (size_t)b * NN * DIMc;
    const bf16_t* efb = ef + (size_t)b * NE * DIMc;
    float h = nfb[n * DIMc + d];
    int cnt = incnt[n];
    for (int t = 0; t < cnt; ++t) {
        int e = inedges[n * 8 + t];
        int s = src[e];
        float m = nfb[s * DIMc + d] + (float)efb[(e & 63) * DIMc + d];
        h += (m > 0.f) ? m : 0.f;
    }
    H[idx] = (bf16_t)h;
}

// pooled[b,d] = mean_n nf[b,n,d]  -> bf16
__global__ void pool_mean(const float* __restrict__ nf, bf16_t* __restrict__ pooled, int total) {
    int idx = blockIdx.x * blockDim.x + threadIdx.x;
    if (idx >= total) return;
    int d = idx % DIMc;
    int b = idx / DIMc;
    const float* p = nf + (size_t)b * NN * DIMc + d;
    float s = 0.f;
#pragma unroll
    for (int n = 0; n < NN; ++n) s += p[n * DIMc];
    pooled[idx] = (bf16_t)(s * (1.f / 32.f));
}

// ------------------------------------------------- bf16 WMMA GEMM: Y = X @ Wt^T + bias
// X  [M,K] row-major bf16, Wt [N,K] row-major bf16, Y [M,N] f32.
// Block tile 128x128, 8 waves (2x4), per wave 4x2 tiles of 16x16, K stepped by 32.
// Tiles staged into double-buffered LDS by the Tensor Data Mover (wave 0 issues the
// 2D descriptors; TENSORcnt + workgroup barrier provide RAW/WAR ordering).
// Requires M%128==0, N%128==0, K%32==0.
__global__ __launch_bounds__(256) void gemm_bf16_wmma(
    const bf16_t* __restrict__ X, const bf16_t* __restrict__ Wt,
    const float* __restrict__ bias, float* __restrict__ Y, int M, int N, int K) {
    __shared__ __align__(128) bf16_t Al[2][128 * 32];
    __shared__ __align__(128) bf16_t Bl[2][128 * 32];

    const int t    = threadIdx.x;
    const int lane = t & 31;
    const int wave = t >> 5;
    const int wm   = wave & 1;   // 0..1 : 64-row group
    const int wn   = wave >> 1;  // 0..3 : 32-col group
    const int m15  = lane & 15;
    const int kh   = lane >> 4;  // lane half selects K sub-range

    const int rowBase = blockIdx.x * 128;
    const int colBase = blockIdx.y * 128;
    const bool tdm_wave = (wave == 0);

    v8f acc[4][2];
#pragma unroll
    for (int i = 0; i < 4; ++i)
#pragma unroll
        for (int j = 0; j < 2; ++j) acc[i][j] = (v8f)(0.f);

#if HAVE_TDM
    // 2D TDM descriptor: 128 rows x 32 bf16, row stride K elements, dense into LDS.
    auto issue_tdm = [&](const bf16_t* gsrc, unsigned lds_off) {
        unsigned long long ga = (unsigned long long)gsrc;
        u32x4 g0;
        g0[0] = 1u;                                        // count=1 (valid user D#)
        g0[1] = lds_off;                                   // lds_addr
        g0[2] = (unsigned)(ga & 0xffffffffu);              // global_addr[31:0]
        g0[3] = (unsigned)((ga >> 32) & 0x01ffffffu)       // global_addr[56:32]
                | (2u << 30);                              // type=2 ("image")
        i32x8 g1;
        g1[0] = 0x00010000;                                // data_size=1 -> 2 bytes
        g1[1] = (int)(32u << 16);                          // tensor_dim0 = 32
        g1[2] = (int)(128u << 16);                         // tensor_dim1 = 128
        g1[3] = (int)(32u << 16);                          // tile_dim0  = 32
        g1[4] = (int)128;                                  // tile_dim1  = 128, tile_dim2 = 0
        g1[5] = (int)K;                                    // tensor_dim0_stride = K elems
        g1[6] = 0;
        g1[7] = 0;
        i32x4 gz4 = {0, 0, 0, 0};                          // groups 2/3: 2D tensor
        i32x8 gz8 = {0, 0, 0, 0, 0, 0, 0, 0};              // trailing group (unused, 2D)
        __builtin_amdgcn_tensor_load_to_lds(g0, g1, gz4, gz4, gz8, 0);
    };

    const int nsteps = K / 32;
    int cur = 0;
    if (tdm_wave) {
        issue_tdm(X  + (size_t)rowBase * K, (unsigned)(unsigned long long)&Al[0][0]);
        issue_tdm(Wt + (size_t)colBase * K, (unsigned)(unsigned long long)&Bl[0][0]);
    }
    for (int s = 0; s < nsteps; ++s) {
        if (tdm_wave) __builtin_amdgcn_s_wait_tensorcnt(0);  // buf[cur] landed in LDS
        __syncthreads();  // publish buf[cur]; all prior LDS reads of buf[cur^1] retired
        if (s + 1 < nsteps && tdm_wave) {
            int k1 = (s + 1) * 32;
            issue_tdm(X  + (size_t)rowBase * K + k1, (unsigned)(unsigned long long)&Al[cur ^ 1][0]);
            issue_tdm(Wt + (size_t)colBase * K + k1, (unsigned)(unsigned long long)&Bl[cur ^ 1][0]);
        }
#else
    const int nsteps = K / 32;
    int cur = 0;
    for (int s = 0; s < nsteps; ++s) {
        int k0 = s * 32;
        __syncthreads();
#pragma unroll
        for (int q = 0; q < 2; ++q) {
            int seg = q * 256 + t;        // 0..511 segments of 8 bf16
            int r   = seg >> 2;
            int c   = (seg & 3) * 8;
            *(int4*)&Al[0][r * 32 + c] = *(const int4*)&X [(size_t)(rowBase + r) * K + k0 + c];
            *(int4*)&Bl[0][r * 32 + c] = *(const int4*)&Wt[(size_t)(colBase + r) * K + k0 + c];
        }
        __syncthreads();
#endif
        union Frag { v16bf v; int4 q[2]; };
        Frag afr[4], bfr[2];
        // A 16x32 frag: lane(m=m15,half=kh): 8 elts at k=kh*8, 8 elts at k=kh*8+16
#pragma unroll
        for (int i = 0; i < 4; ++i) {
            int row = wm * 64 + i * 16 + m15;
            afr[i].q[0] = *(const int4*)&Al[cur][row * 32 + kh * 8];
            afr[i].q[1] = *(const int4*)&Al[cur][row * 32 + kh * 8 + 16];
        }
        // B 32x16 frag: lane(n=m15,half=kh): 8 elts at k=kh*8, 8 elts at k=kh*8+16
#pragma unroll
        for (int j = 0; j < 2; ++j) {
            int col = wn * 32 + j * 16 + m15;
            bfr[j].q[0] = *(const int4*)&Bl[cur][col * 32 + kh * 8];
            bfr[j].q[1] = *(const int4*)&Bl[cur][col * 32 + kh * 8 + 16];
        }
#pragma unroll
        for (int i = 0; i < 4; ++i)
#pragma unroll
            for (int j = 0; j < 2; ++j)
                acc[i][j] = __builtin_amdgcn_wmma_f32_16x16x32_bf16(
                    false, afr[i].v, false, bfr[j].v, (short)0, acc[i][j], false, false);
#if HAVE_TDM
        cur ^= 1;
#endif
    }

    // epilogue: D lane(n=m15) holds rows v + 8*kh
#pragma unroll
    for (int i = 0; i < 4; ++i)
#pragma unroll
        for (int j = 0; j < 2; ++j) {
            int col  = colBase + wn * 32 + j * 16 + m15;
            float bv = bias ? bias[col] : 0.f;
#pragma unroll
            for (int v = 0; v < 8; ++v) {
                int row = rowBase + wm * 64 + i * 16 + kh * 8 + v;
                Y[(size_t)row * N + col] = acc[i][j][v] + bv;
            }
        }
}

// ---------------------------------------------------------------- launcher

extern "C" void kernel_launch(void* const* d_in, const int* in_sizes, int n_in,
                              void* d_out, int out_size, void* d_ws, size_t ws_size,
                              hipStream_t stream) {
    const float* x    = (const float*)d_in[0];
    const float* nfe  = (const float*)d_in[1];
    const float* efe  = (const float*)d_in[2];
    const float* W    = (const float*)d_in[3];
    const float* bvec = (const float*)d_in[4];
    const float* Wout = (const float*)d_in[5];
    const float* bout = (const float*)d_in[6];
    const int*   src  = (const int*)d_in[7];
    const int*   dstv = (const int*)d_in[8];

    char* ws = (char*)d_ws;
    size_t off = 0;
    auto alloc = [&](size_t bytes) -> void* {
        void* p = ws + off;
        off += (bytes + 255) & ~(size_t)255;
        return p;
    };
    float*  nf      = (float*) alloc((size_t)Bsz * NN * DIMc * 4);   //  96 MiB master f32
    bf16_t* ef      = (bf16_t*)alloc((size_t)Bsz * NE * DIMc * 2);   //  96 MiB
    bf16_t* H       = (bf16_t*)alloc((size_t)Bsz * NN * DIMc * 2);   //  48 MiB
    bf16_t* Wbf     = (bf16_t*)alloc((size_t)DEPTHc * DIMc * DIMc * 2);
    bf16_t* Woutbf  = (bf16_t*)alloc((size_t)OUTD * DIMc * 2);
    bf16_t* pooled  = (bf16_t*)alloc((size_t)Bsz * DIMc * 2);
    int*    incnt   = (int*)   alloc(NN * 4);
    int*    inedges = (int*)   alloc(NN * 8 * 4);

    const int TPB = 256;
    // weight conversion (f32 -> bf16)
    {
        int n1 = DEPTHc * DIMc * DIMc;
        cvt_f32_bf16<<<(n1 + TPB - 1) / TPB, TPB, 0, stream>>>(W, Wbf, n1);
        int n2 = OUTD * DIMc;
        cvt_f32_bf16<<<(n2 + TPB - 1) / TPB, TPB, 0, stream>>>(Wout, Woutbf, n2);
    }
    build_csr<<<1, 32, 0, stream>>>(dstv, incnt, inedges);

    {
        int total = Bsz * (NN + NE) * DIMc;
        init_feats<<<(total + TPB - 1) / TPB, TPB, 0, stream>>>(x, nfe, efe, nf, ef, total);
    }

    const int Mrows = Bsz * NN;  // 32768
    for (int l = 0; l < DEPTHc; ++l) {
        int total = Mrows * DIMc;
        msg_agg<<<(total + TPB - 1) / TPB, TPB, 0, stream>>>(nf, ef, src, incnt, inedges, H, total);
        gemm_bf16_wmma<<<dim3(Mrows / 128, DIMc / 128), 256, 0, stream>>>(
            H, Wbf + (size_t)l * DIMc * DIMc, bvec + (size_t)l * DIMc, nf, Mrows, DIMc, DIMc);
    }

    {
        int total = Bsz * DIMc;
        pool_mean<<<(total + TPB - 1) / TPB, TPB, 0, stream>>>(nf, pooled, total);
    }
    gemm_bf16_wmma<<<dim3(Bsz / 128, OUTD / 128), 256, 0, stream>>>(
        pooled, Woutbf, bout, (float*)d_out, Bsz, OUTD, DIMc);
}
